// Speller_58987080843784
// MI455X (gfx1250) — matrix operand order repaired and checked by reading, not compiled, not run on hardware
//
#include <hip/hip_runtime.h>
#include <hip/hip_bf16.h>

typedef __attribute__((ext_vector_type(16))) __bf16 v16bf;
typedef __attribute__((ext_vector_type(8)))  __bf16 v8bf;
typedef __attribute__((ext_vector_type(8)))  float  v8f;

#define B_      64
#define TE_     1600
#define EH_     512
#define H_      512
#define P_      512
#define V_      42
#define MAXLEN_ 40

// ---------------------------------------------------------------------------
// WMMA fragment loaders (bf16, 16x16x32).
// A-matrix 16x32 (MxK): lanes 0-15 row M=lane, K = {kk+0..7, kk+16..23};
//                       lanes 16-31 row M=lane-16, K = {kk+8..15, kk+24..31}.
// B-operand holds column n of B = row n of W[N,K]: lanes 0-15 n=lane,
//   K = {kk+0..7, kk+16..23}; lanes 16-31 n=lane-16, K = {kk+8..15, kk+24..31}.
// ---------------------------------------------------------------------------
__device__ inline v16bf load_frag_a_bf16(const __bf16* tile, int lda, int kk, int lane) {
    int r    = lane & 15;
    int koff = kk + ((lane >> 4) << 3);
    const __bf16* p = tile + (size_t)r * lda + koff;
    v8bf lo = *reinterpret_cast<const v8bf*>(p);
    v8bf hi = *reinterpret_cast<const v8bf*>(p + 16);
    v16bf f;
#pragma unroll
    for (int i = 0; i < 8; ++i) { f[i] = lo[i]; f[8 + i] = hi[i]; }
    return f;
}

__device__ inline v16bf load_frag_a_f32(const float* tile, int lda, int kk, int lane) {
    int r    = lane & 15;
    int koff = kk + ((lane >> 4) << 3);
    const float* p = tile + (size_t)r * lda + koff;
    v16bf f;
#pragma unroll
    for (int i = 0; i < 8; ++i) { f[i] = (__bf16)p[i]; f[8 + i] = (__bf16)p[16 + i]; }
    return f;
}

__device__ inline v16bf load_frag_a(const __bf16* t, int lda, int kk, int lane) { return load_frag_a_bf16(t, lda, kk, lane); }
__device__ inline v16bf load_frag_a(const float*  t, int lda, int kk, int lane) { return load_frag_a_f32 (t, lda, kk, lane); }

__device__ inline v16bf load_frag_b(const __bf16* Wtile, int ldw, int kk, int lane) {
    int n    = lane & 15;
    int koff = kk + ((lane >> 4) << 3);
    const __bf16* p = Wtile + (size_t)n * ldw + koff;
    v8bf lo = *reinterpret_cast<const v8bf*>(p);
    v8bf hi = *reinterpret_cast<const v8bf*>(p + 16);
    v16bf f;
#pragma unroll
    for (int i = 0; i < 8; ++i) { f[i] = lo[i]; f[8 + i] = hi[i]; }
    return f;
}

// ---------------------------------------------------------------------------
// Generic WMMA GEMM:  C[M,N] = act(A[M,K] * W[N,K]^T + bias[N])
// Block = 256 threads = 8 waves laid out 4(M) x 2(N): block tile 64M x 128N.
// Each wave: 16M x 64N = 4 accumulators, K stepped by 32.
// Grid-sized exactly: M%64==0, N%128==0, K%32==0 for every call below.
// ---------------------------------------------------------------------------
template <typename TA>
__global__ void gemm_wmma(const TA* __restrict__ A, int lda,
                          const __bf16* __restrict__ W, int ldw,
                          const float* __restrict__ bias,
                          float* __restrict__ Cf, __bf16* __restrict__ Cbf,
                          int M, int N, int K, int act) {
    const int lane = threadIdx.x & 31;
    const int wave = threadIdx.x >> 5;
    const int m0 = blockIdx.x * 64 + (wave & 3) * 16;
    const int n0 = blockIdx.y * 128 + (wave >> 2) * 64;
    if (m0 >= M || n0 >= N) return;   // wave-uniform

    v8f acc[4] = {};
    for (int kk = 0; kk < K; kk += 32) {
        v16bf a = load_frag_a(A + (size_t)m0 * lda, lda, kk, lane);
#pragma unroll
        for (int j = 0; j < 4; ++j) {
            v16bf b = load_frag_b(W + (size_t)(n0 + 16 * j) * ldw, ldw, kk, lane);
            acc[j] = __builtin_amdgcn_wmma_f32_16x16x32_bf16(
                false, a, false, b, (short)0, acc[j], false, false);
        }
    }
    // C/D layout: lanes 0-15 -> M = m0+r (VGPR r), lanes 16-31 -> M = m0+8+r; N = n0+16j+(lane&15)
    const int nbase = n0 + (lane & 15);
    const int mrow0 = m0 + ((lane >> 4) << 3);
#pragma unroll
    for (int j = 0; j < 4; ++j) {
        int n = nbase + 16 * j;
        float bs = bias ? bias[n] : 0.0f;
#pragma unroll
        for (int r = 0; r < 8; ++r) {
            float v = acc[j][r] + bs;
            if (act) v = tanhf(v);
            size_t idx = (size_t)(mrow0 + r) * N + n;
            if (Cf)  Cf[idx]  = v;
            if (Cbf) Cbf[idx] = (__bf16)v;
        }
    }
}

// ---------------------------------------------------------------------------
// Elementwise / small kernels
// ---------------------------------------------------------------------------
__global__ void convert_rows(const float* __restrict__ src, __bf16* __restrict__ dst,
                             int R, int C, int dld, int coff) {
    int i = blockIdx.x * 256 + threadIdx.x;
    if (i >= R * C) return;
    int r = i / C, c = i - r * C;
    dst[(size_t)r * dld + coff + c] = (__bf16)src[i];
}

__global__ void add_vec(const float* a, const float* b, float* o, int n) {
    int i = blockIdx.x * 256 + threadIdx.x;
    if (i < n) o[i] = a[i] + b[i];
}

__global__ void copy_vec(const float* a, float* o, int n) {
    int i = blockIdx.x * 256 + threadIdx.x;
    if (i < n) o[i] = a[i];
}

__global__ void zero_u32(unsigned* p, int n) {
    int i = blockIdx.x * 256 + threadIdx.x;
    if (i < n) p[i] = 0u;
}

// A1[:, :512] = emb_bf[char[b]]
__global__ void embed_kernel(const int* __restrict__ ch, const __bf16* __restrict__ emb,
                             __bf16* __restrict__ A1) {
    int i = blockIdx.x * 256 + threadIdx.x;   // 64*512
    int b = i >> 9, j = i & 511;
    A1[(size_t)b * (H_ + P_ + H_) + j] = emb[(size_t)ch[b] * H_ + j];
}

__device__ inline float sigmoidf(float x) { return 1.0f / (1.0f + expf(-x)); }

// gates [64,2048] (i|f|g|o), c in/out f32 [64,512]; h written as bf16 to two dests
__global__ void lstm_update(const float* __restrict__ gates, float* __restrict__ c,
                            __bf16* __restrict__ d0, int ld0,
                            __bf16* __restrict__ d1, int ld1) {
    int i = blockIdx.x * 256 + threadIdx.x;   // 64*512
    int b = i >> 9, j = i & 511;
    const float* g = gates + (size_t)b * (4 * H_);
    float ig = sigmoidf(g[j]);
    float fg = sigmoidf(g[H_ + j]);
    float gg = tanhf(g[2 * H_ + j]);
    float og = sigmoidf(g[3 * H_ + j]);
    float cn = fg * c[i] + ig * gg;
    float hn = og * tanhf(cn);
    c[i] = cn;
    __bf16 hb = (__bf16)hn;
    d0[(size_t)b * ld0 + j] = hb;
    d1[(size_t)b * ld1 + j] = hb;
}

// energy[b,t] = mask ? dot(q[b,:], kv[b,t,0:512]) : -1e9 ; one wave per (b,t)
// kv row stride 1024 (keys | values fused)
__global__ void energy_kernel(const __bf16* __restrict__ q, const __bf16* __restrict__ kv,
                              const int* __restrict__ lens, float* __restrict__ energy) {
    int b    = blockIdx.x;
    int wave = threadIdx.x >> 5, lane = threadIdx.x & 31;
    int t    = blockIdx.y * 8 + wave;
    const __bf16* kp = kv + ((size_t)b * TE_ + t) * (P_ * 2) + lane * 16;
    const __bf16* qp = q + (size_t)b * P_ + lane * 16;
    float acc = 0.0f;
#pragma unroll
    for (int i = 0; i < 16; ++i) acc += (float)kp[i] * (float)qp[i];
#pragma unroll
    for (int off = 16; off > 0; off >>= 1) acc += __shfl_xor(acc, off);
    if (lane == 0) energy[(size_t)b * TE_ + t] = (t < lens[b]) ? acc : -1e9f;
}

// block-per-batch softmax over TE, writing attention_plot (doubles as weights)
__global__ void softmax_kernel(const float* __restrict__ energy,
                               float* __restrict__ attnplot, int step) {
    __shared__ float red[256];
    int b = blockIdx.x, tid = threadIdx.x;
    const float* e = energy + (size_t)b * TE_;
    float mx = -INFINITY;
    for (int t = tid; t < TE_; t += 256) mx = fmaxf(mx, e[t]);
    red[tid] = mx; __syncthreads();
    for (int s = 128; s > 0; s >>= 1) { if (tid < s) red[tid] = fmaxf(red[tid], red[tid + s]); __syncthreads(); }
    mx = red[0]; __syncthreads();
    float sm = 0.0f;
    for (int t = tid; t < TE_; t += 256) sm += expf(e[t] - mx);
    red[tid] = sm; __syncthreads();
    for (int s = 128; s > 0; s >>= 1) { if (tid < s) red[tid] += red[tid + s]; __syncthreads(); }
    float inv = 1.0f / red[0];
    float* out = attnplot + ((size_t)b * MAXLEN_ + step) * TE_;
    for (int t = tid; t < TE_; t += 256) out[t] = expf(e[t] - mx) * inv;
}

// ctx[b,p] = sum_t attn[b,t] * kv[b,t,512+p]; write bf16 into A1 ctx slot + feat slot
__global__ void ctx_kernel(const float* __restrict__ attnplot, const __bf16* __restrict__ kv,
                           int step, __bf16* __restrict__ d0, int ld0,
                           __bf16* __restrict__ d1, int ld1) {
    int b = blockIdx.x;
    int p = blockIdx.y * 256 + threadIdx.x;   // 512 = 2 * 256
    const float*  a = attnplot + ((size_t)b * MAXLEN_ + step) * TE_;
    const __bf16* v = kv + (size_t)b * TE_ * (P_ * 2) + P_ + p;
    float acc = 0.0f;
    for (int t = 0; t < TE_; ++t) acc += a[t] * (float)v[(size_t)t * (P_ * 2)];
    __bf16 cb = (__bf16)acc;
    d0[(size_t)b * ld0 + p] = cb;
    d1[(size_t)b * ld1 + p] = cb;
}

// pred[b,v] = dot(o_bf[b,:], emb[v,:]) + b_char[v]; write logits; argmax -> char
__global__ void pred_kernel(const __bf16* __restrict__ o_bf, const __bf16* __restrict__ emb,
                            const float* __restrict__ bchar, float* __restrict__ predout,
                            int step, int* __restrict__ ch) {
    __shared__ float vals[64];
    __shared__ int   idxs[64];
    int b = blockIdx.x, v = threadIdx.x;
    float acc = -INFINITY;
    if (v < V_) {
        const __bf16* o = o_bf + (size_t)b * H_;
        const __bf16* e = emb + (size_t)v * H_;
        float s = bchar[v];
        for (int k = 0; k < H_; ++k) s += (float)o[k] * (float)e[k];
        predout[((size_t)b * MAXLEN_ + step) * V_ + v] = s;
        acc = s;
    }
    vals[v] = acc; idxs[v] = v; __syncthreads();
    for (int s = 32; s > 0; s >>= 1) {
        if (v < s) {
            int o2 = v + s;
            if (vals[o2] > vals[v] || (vals[o2] == vals[v] && idxs[o2] < idxs[v])) {
                vals[v] = vals[o2]; idxs[v] = idxs[o2];
            }
        }
        __syncthreads();
    }
    if (v == 0) ch[b] = idxs[0];
}

// ---------------------------------------------------------------------------
extern "C" void kernel_launch(void* const* d_in, const int* in_sizes, int n_in,
                              void* d_out, int out_size, void* d_ws, size_t ws_size,
                              hipStream_t stream) {
    const float* enc    = (const float*)d_in[0];
    const int*   lens   = (const int*)  d_in[1];
    const float* emb    = (const float*)d_in[2];
    const float* W_ih1  = (const float*)d_in[3];
    const float* W_hh1  = (const float*)d_in[4];
    const float* b_ih1  = (const float*)d_in[5];
    const float* b_hh1  = (const float*)d_in[6];
    const float* W_ih2  = (const float*)d_in[7];
    const float* W_hh2  = (const float*)d_in[8];
    const float* b_ih2  = (const float*)d_in[9];
    const float* b_hh2  = (const float*)d_in[10];
    const float* Wk     = (const float*)d_in[11];
    const float* bk     = (const float*)d_in[12];
    const float* Wv     = (const float*)d_in[13];
    const float* bv     = (const float*)d_in[14];
    const float* Wq     = (const float*)d_in[15];
    const float* bq     = (const float*)d_in[16];
    const float* Wo     = (const float*)d_in[17];
    const float* bo     = (const float*)d_in[18];
    const float* bchar  = (const float*)d_in[19];

    float* pred_out = (float*)d_out;                              // [64,40,42]
    float* attnplot = (float*)d_out + (size_t)B_ * MAXLEN_ * V_;  // [64,40,1600]

    // ---- workspace layout (256B aligned regions) ----
    char*  ws  = (char*)d_ws;
    size_t off = 0;
    auto alloc = [&](size_t bytes) { size_t o = off; off = (off + bytes + 255) & ~(size_t)255; return o; };
    const int KC1 = H_ + P_ + H_;   // 1536: A1 = [emb | ctx | h1]
    const int KC2 = H_ + H_;        // 1024: A2 = [h1 | h2]

    __bf16* kv_bf    = (__bf16*)(ws + alloc((size_t)B_ * TE_ * (P_ * 2) * 2)); // [102400, 1024] keys|values
    __bf16* W1cat    = (__bf16*)(ws + alloc((size_t)4 * H_ * KC1 * 2));
    __bf16* W2cat    = (__bf16*)(ws + alloc((size_t)4 * H_ * KC2 * 2));
    __bf16* Wq_bf    = (__bf16*)(ws + alloc((size_t)P_ * H_ * 2));
    __bf16* Wo_bf    = (__bf16*)(ws + alloc((size_t)H_ * (H_ + P_) * 2));
    __bf16* Wkv_bf   = (__bf16*)(ws + alloc((size_t)(2 * P_) * EH_ * 2));      // [1024, 512] Wk|Wv rows
    __bf16* emb_bf   = (__bf16*)(ws + alloc((size_t)V_ * H_ * 2));
    float*  bias1    = (float*) (ws + alloc((size_t)4 * H_ * 4));
    float*  bias2    = (float*) (ws + alloc((size_t)4 * H_ * 4));
    float*  bias_kv  = (float*) (ws + alloc((size_t)2 * P_ * 4));
    __bf16* A1       = (__bf16*)(ws + alloc((size_t)B_ * KC1 * 2));
    __bf16* A2       = (__bf16*)(ws + alloc((size_t)B_ * KC2 * 2));
    __bf16* feat     = (__bf16*)(ws + alloc((size_t)B_ * (H_ + P_) * 2));
    float*  c1       = (float*) (ws + alloc((size_t)B_ * H_ * 4));
    float*  c2       = (float*) (ws + alloc((size_t)B_ * H_ * 4));
    float*  gates1   = (float*) (ws + alloc((size_t)B_ * 4 * H_ * 4));
    float*  gates2   = (float*) (ws + alloc((size_t)B_ * 4 * H_ * 4));
    __bf16* q_bf     = (__bf16*)(ws + alloc((size_t)B_ * P_ * 2));
    __bf16* o_bf     = (__bf16*)(ws + alloc((size_t)B_ * H_ * 2));
    float*  energy   = (float*) (ws + alloc((size_t)B_ * TE_ * 4));
    int*    chbuf    = (int*)   (ws + alloc((size_t)B_ * 4));
    (void)ws_size; (void)in_sizes; (void)n_in; (void)out_size;

    // ---- one-time prep: bf16 weights, fused biases, zeroed state ----
    auto cvt = [&](const float* s, __bf16* d, int R, int C, int dld, int coff) {
        convert_rows<<<(R * C + 255) / 256, 256, 0, stream>>>(s, d, R, C, dld, coff);
    };
    cvt(W_ih1, W1cat, 4 * H_, H_ + P_, KC1, 0);
    cvt(W_hh1, W1cat, 4 * H_, H_,      KC1, H_ + P_);
    cvt(W_ih2, W2cat, 4 * H_, H_,      KC2, 0);
    cvt(W_hh2, W2cat, 4 * H_, H_,      KC2, H_);
    cvt(Wq,  Wq_bf,        P_, H_,      H_,      0);
    cvt(Wo,  Wo_bf,        H_, H_ + P_, H_ + P_, 0);
    cvt(Wk,  Wkv_bf,       P_, EH_,     EH_,     0);   // rows 0..511
    cvt(Wv,  Wkv_bf + (size_t)P_ * EH_, P_, EH_, EH_, 0); // rows 512..1023
    cvt(emb, emb_bf,       V_, H_,      H_,      0);
    add_vec<<<(4 * H_ + 255) / 256, 256, 0, stream>>>(b_ih1, b_hh1, bias1, 4 * H_);
    add_vec<<<(4 * H_ + 255) / 256, 256, 0, stream>>>(b_ih2, b_hh2, bias2, 4 * H_);
    copy_vec<<<(P_ + 255) / 256, 256, 0, stream>>>(bk, bias_kv,      P_);
    copy_vec<<<(P_ + 255) / 256, 256, 0, stream>>>(bv, bias_kv + P_, P_);

    auto zero = [&](void* p, size_t bytes) {
        int n = (int)(bytes / 4);
        zero_u32<<<(n + 255) / 256, 256, 0, stream>>>((unsigned*)p, n);
    };
    zero(A1,   (size_t)B_ * KC1 * 2);          // emb/ctx/h1 slots = 0
    zero(A2,   (size_t)B_ * KC2 * 2);          // h1/h2 slots = 0
    zero(feat, (size_t)B_ * (H_ + P_) * 2);
    zero(c1,   (size_t)B_ * H_ * 4);
    zero(c2,   (size_t)B_ * H_ * 4);
    zero(chbuf,(size_t)B_ * 4);                // SOS = 0

    // ---- keys|values fused: one WMMA GEMM, encoder f32 read exactly once ----
    {
        dim3 grid((B_ * TE_) / 64, (2 * P_) / 128);  // (1600, 8)
        gemm_wmma<float><<<grid, 256, 0, stream>>>(enc, EH_, Wkv_bf, EH_, bias_kv,
                                                   nullptr, kv_bf, B_ * TE_, 2 * P_, EH_, 0);
    }

    // ---- 40 sequential decode steps ----
    for (int step = 0; step < MAXLEN_; ++step) {
        embed_kernel<<<(B_ * H_) / 256, 256, 0, stream>>>(chbuf, emb_bf, A1);

        gemm_wmma<__bf16><<<dim3(1, (4 * H_) / 128), 256, 0, stream>>>(
            A1, KC1, W1cat, KC1, bias1, gates1, nullptr, B_, 4 * H_, KC1, 0);

        lstm_update<<<(B_ * H_) / 256, 256, 0, stream>>>(
            gates1, c1, A1 + (H_ + P_), KC1, A2, KC2);        // h1 -> A1 slot (next step) + A2[:, :512]

        gemm_wmma<__bf16><<<dim3(1, (4 * H_) / 128), 256, 0, stream>>>(
            A2, KC2, W2cat, KC2, bias2, gates2, nullptr, B_, 4 * H_, KC2, 0);

        lstm_update<<<(B_ * H_) / 256, 256, 0, stream>>>(
            gates2, c2, A2 + H_, KC2, feat, H_ + P_);         // h2 -> A2[:, 512:] + feat[:, :512]

        gemm_wmma<__bf16><<<dim3(1, P_ / 128), 256, 0, stream>>>(
            A2 + H_, KC2, Wq_bf, H_, bq, nullptr, q_bf, B_, P_, H_, 0);

        energy_kernel<<<dim3(B_, TE_ / 8), 256, 0, stream>>>(q_bf, kv_bf, lens, energy);
        softmax_kernel<<<B_, 256, 0, stream>>>(energy, attnplot, step);

        ctx_kernel<<<dim3(B_, P_ / 256), 256, 0, stream>>>(
            attnplot, kv_bf, step, A1 + H_, KC1, feat + H_, H_ + P_);

        gemm_wmma<__bf16><<<dim3(1, H_ / 128), 256, 0, stream>>>(
            feat, H_ + P_, Wo_bf, H_ + P_, bo, nullptr, o_bf, B_, H_, H_ + P_, 1);

        pred_kernel<<<B_, 64, 0, stream>>>(o_bf, emb_bf, bchar, pred_out, step, chbuf);
    }
}
